// HeteroLinear_53489522704783
// MI455X (gfx1250) — matrix operand order repaired and checked by reading, not compiled, not run on hardware
//
#include <hip/hip_runtime.h>

// ---------------------------------------------------------------------------
// HeteroLinear on MI455X (gfx1250, wave32)
//   out[n] = x[n] @ W[type[n]].T + b[type[n]]
// Strategy: bucket rows by type (hist + scan + scatter into d_ws), then a
// grouped GEMM using v_wmma_f32_16x16x32_bf16 with bf16 hi/lo split-precision
// (3 WMMAs per tile: hi*hi + hi*lo + lo*hi) so accuracy stays ~fp32 while the
// math runs on the bf16 matrix pipe. Memory-bound: ~412 MB HBM => ~18 us.
// ---------------------------------------------------------------------------

typedef __attribute__((ext_vector_type(16))) __bf16 v16bf;
typedef __attribute__((ext_vector_type(8)))  __bf16 v8bf;
typedef __attribute__((ext_vector_type(8)))  float  v8f;

#define NTYPES 16
#define INCH   256
#define OUTCH  256
#define TILE_M 64
#define KCHUNK 32
#define SXROW  (KCHUNK + 8)   // 40 bf16 elems -> 80B row stride (16B aligned)
#define SWROW  (KCHUNK + 8)

// d_ws int layout: counts[16] @0, cursors[16] @16, seg[16] @32, perm[N] @48

static __device__ __forceinline__ unsigned short bf16_trunc_bits(float f) {
    unsigned u = __builtin_bit_cast(unsigned, f);
    return (unsigned short)(u >> 16);
}
static __device__ __forceinline__ float trunc_hi(float f) {
    unsigned u = __builtin_bit_cast(unsigned, f) & 0xFFFF0000u;
    return __builtin_bit_cast(float, u);
}
static __device__ __forceinline__ unsigned short bf16_rne_bits(float f) {
    unsigned u = __builtin_bit_cast(unsigned, f);
    unsigned r = u + 0x7FFFu + ((u >> 16) & 1u);
    return (unsigned short)(r >> 16);
}

__global__ void hl_zero(int* ws) {
    int t = threadIdx.x;
    if (t < 32) ws[t] = 0;   // counts + cursors
}

__global__ void hl_hist(const int* __restrict__ tvec, int n, int* ws) {
    int i = blockIdx.x * blockDim.x + threadIdx.x;
    if (i < n) atomicAdd(&ws[tvec[i] & (NTYPES - 1)], 1);
}

__global__ void hl_scan(int* ws) {
    if (threadIdx.x == 0) {
        int s = 0;
        for (int t = 0; t < NTYPES; ++t) {
            int c = ws[t];       // counts
            ws[32 + t] = s;      // seg start (immutable)
            ws[16 + t] = s;      // cursor
            s += c;
        }
    }
}

__global__ void hl_scatter(const int* __restrict__ tvec, int n, int* ws) {
    int i = blockIdx.x * blockDim.x + threadIdx.x;
    if (i < n) {
        int t = tvec[i] & (NTYPES - 1);
        int pos = atomicAdd(&ws[16 + t], 1);
        ws[48 + pos] = i;        // perm
    }
}

// ---------------------------------------------------------------------------
// Grouped GEMM. Block = 256 threads = 8 waves, tile = 64 rows x 256 cols.
// Wave (mrow 0..1, ncol 0..3) computes 32x64 via 2x4 accumulators of 16x16.
// ---------------------------------------------------------------------------
__global__ __launch_bounds__(256)
void hl_gemm(const float* __restrict__ x, const float* __restrict__ W,
             const float* __restrict__ bias, const int* __restrict__ ws,
             float* __restrict__ out) {
    const int* counts = ws;
    const int* seg    = ws + 32;
    const int* perm   = ws + 48;

    // Map flat tile id -> (type, tile-within-segment). Uniform scalar walk.
    int tid = blockIdx.x;
    int type = -1, tileInSeg = 0;
    for (int t = 0; t < NTYPES; ++t) {
        int len = counts[t];
        int ntile = (len + TILE_M - 1) / TILE_M;
        if (tid < ntile) { type = t; tileInSeg = tid; break; }
        tid -= ntile;
    }
    if (type < 0) return;

    const int segBase  = seg[type];
    const int segLen   = counts[type];
    const int rowBase  = tileInSeg * TILE_M;
    const int rowsHere = (segLen - rowBase) < TILE_M ? (segLen - rowBase) : TILE_M;

    __shared__ unsigned short xhi[TILE_M * SXROW];
    __shared__ unsigned short xlo[TILE_M * SXROW];
    __shared__ unsigned short whi[OUTCH * SWROW];
    __shared__ unsigned short wlo[OUTCH * SWROW];

    const int lane      = threadIdx.x & 31;
    const int wave      = threadIdx.x >> 5;
    const int laneMod16 = lane & 15;
    const int laneHi    = lane >> 4;        // 0 or 1
    const int mrow      = wave >> 2;        // 0..1
    const int ncol      = wave & 3;         // 0..3

    v8f acc[2][4];
#pragma unroll
    for (int mt = 0; mt < 2; ++mt)
#pragma unroll
        for (int nt = 0; nt < 4; ++nt)
#pragma unroll
            for (int e = 0; e < 8; ++e) acc[mt][nt][e] = 0.0f;

    const float* Wt = W + (size_t)type * (OUTCH * INCH);

    for (int kc = 0; kc < INCH / KCHUNK; ++kc) {
        __syncthreads();   // previous chunk's reads done before overwrite

        // ---- stage x chunk: 64 rows x 32 k, gathered through perm --------
#pragma unroll
        for (int j = 0; j < 2; ++j) {
            int idx = threadIdx.x + j * 256;      // 0..511
            int r   = idx >> 3;                   // row in tile
            int c4  = idx & 7;                    // float4 within chunk
            int rc  = r < rowsHere ? r : rowsHere - 1;
            int grow = perm[segBase + rowBase + rc];
            const float4 v = *(const float4*)(x + (size_t)grow * INCH + kc * KCHUNK + c4 * 4);
            int off = r * SXROW + c4 * 4;
            float e0 = v.x, e1 = v.y, e2 = v.z, e3 = v.w;
            xhi[off + 0] = bf16_trunc_bits(e0); xlo[off + 0] = bf16_rne_bits(e0 - trunc_hi(e0));
            xhi[off + 1] = bf16_trunc_bits(e1); xlo[off + 1] = bf16_rne_bits(e1 - trunc_hi(e1));
            xhi[off + 2] = bf16_trunc_bits(e2); xlo[off + 2] = bf16_rne_bits(e2 - trunc_hi(e2));
            xhi[off + 3] = bf16_trunc_bits(e3); xlo[off + 3] = bf16_rne_bits(e3 - trunc_hi(e3));
        }

        // ---- stage W chunk: 256 out-ch rows x 32 k ------------------------
#pragma unroll
        for (int j = 0; j < 8; ++j) {
            int idx = threadIdx.x + j * 256;      // 0..2047
            int nn  = idx >> 3;
            int c4  = idx & 7;
            const float4 v = *(const float4*)(Wt + (size_t)nn * INCH + kc * KCHUNK + c4 * 4);
            int off = nn * SWROW + c4 * 4;
            float e0 = v.x, e1 = v.y, e2 = v.z, e3 = v.w;
            whi[off + 0] = bf16_trunc_bits(e0); wlo[off + 0] = bf16_rne_bits(e0 - trunc_hi(e0));
            whi[off + 1] = bf16_trunc_bits(e1); wlo[off + 1] = bf16_rne_bits(e1 - trunc_hi(e1));
            whi[off + 2] = bf16_trunc_bits(e2); wlo[off + 2] = bf16_rne_bits(e2 - trunc_hi(e2));
            whi[off + 3] = bf16_trunc_bits(e3); wlo[off + 3] = bf16_rne_bits(e3 - trunc_hi(e3));
        }

        __syncthreads();

        // ---- A fragments (16x32 bf16, ISA lane layout) --------------------
        // lanes 0-15: elems 0..7 = K 0..7,  elems 8..15 = K 16..23
        // lanes16-31: elems 0..7 = K 8..15, elems 8..15 = K 24..31
        v16bf a_hi[2], a_lo[2];
        const int kbA = laneHi * 8;
#pragma unroll
        for (int mt = 0; mt < 2; ++mt) {
            int m = mrow * 32 + mt * 16 + laneMod16;
            const v8bf h0 = *(const v8bf*)&xhi[m * SXROW + kbA];
            const v8bf h1 = *(const v8bf*)&xhi[m * SXROW + kbA + 16];
            const v8bf l0 = *(const v8bf*)&xlo[m * SXROW + kbA];
            const v8bf l1 = *(const v8bf*)&xlo[m * SXROW + kbA + 16];
#pragma unroll
            for (int e = 0; e < 8; ++e) {
                a_hi[mt][e] = h0[e]; a_hi[mt][8 + e] = h1[e];
                a_lo[mt][e] = l0[e]; a_lo[mt][8 + e] = l1[e];
            }
        }

        // ---- B fragments (32x16 bf16): same K layout as A ------------------
        const int kbB = laneHi * 8;
#pragma unroll
        for (int nt = 0; nt < 4; ++nt) {
            int n = ncol * 64 + nt * 16 + laneMod16;
            const v8bf h0 = *(const v8bf*)&whi[n * SWROW + kbB];
            const v8bf h1 = *(const v8bf*)&whi[n * SWROW + kbB + 16];
            const v8bf l0 = *(const v8bf*)&wlo[n * SWROW + kbB];
            const v8bf l1 = *(const v8bf*)&wlo[n * SWROW + kbB + 16];
            v16bf b_hi, b_lo;
#pragma unroll
            for (int e = 0; e < 8; ++e) {
                b_hi[e] = h0[e]; b_hi[8 + e] = h1[e];
                b_lo[e] = l0[e]; b_lo[8 + e] = l1[e];
            }
#pragma unroll
            for (int mt = 0; mt < 2; ++mt) {
                // split-precision accumulate: lo*hi + hi*lo + hi*hi
                acc[mt][nt] = __builtin_amdgcn_wmma_f32_16x16x32_bf16(
                    false, a_lo[mt], false, b_hi, (short)0, acc[mt][nt], false, false);
                acc[mt][nt] = __builtin_amdgcn_wmma_f32_16x16x32_bf16(
                    false, a_hi[mt], false, b_lo, (short)0, acc[mt][nt], false, false);
                acc[mt][nt] = __builtin_amdgcn_wmma_f32_16x16x32_bf16(
                    false, a_hi[mt], false, b_hi, (short)0, acc[mt][nt], false, false);
            }
        }
    }

    // ---- bias + scatter store --------------------------------------------
    // C/D layout: VGPR r, lanes 0-15: (M=r, N=lane); lanes 16-31: (M=r+8, N=lane-16)
#pragma unroll
    for (int nt = 0; nt < 4; ++nt) {
        int col = ncol * 64 + nt * 16 + laneMod16;
        float bv = bias[type * OUTCH + col];
#pragma unroll
        for (int mt = 0; mt < 2; ++mt) {
#pragma unroll
            for (int r = 0; r < 8; ++r) {
                int ml = mrow * 32 + mt * 16 + laneHi * 8 + r;
                if (ml < rowsHere) {
                    int grow = perm[segBase + rowBase + ml];
                    out[(size_t)grow * OUTCH + col] = acc[mt][nt][r] + bv;
                }
            }
        }
    }
}

extern "C" void kernel_launch(void* const* d_in, const int* in_sizes, int n_in,
                              void* d_out, int out_size, void* d_ws, size_t ws_size,
                              hipStream_t stream) {
    const float* x    = (const float*)d_in[0];
    const int*   tvec = (const int*)d_in[1];
    const float* W    = (const float*)d_in[2];
    const float* bias = (const float*)d_in[3];
    float* out = (float*)d_out;
    int* ws = (int*)d_ws;

    const int nrows = in_sizes[1];          // one type id per row
    const int nb    = (nrows + 255) / 256;

    hl_zero<<<1, 64, 0, stream>>>(ws);
    hl_hist<<<nb, 256, 0, stream>>>(tvec, nrows, ws);
    hl_scan<<<1, 32, 0, stream>>>(ws);
    hl_scatter<<<nb, 256, 0, stream>>>(tvec, nrows, ws);

    const int maxTiles = (nrows + TILE_M - 1) / TILE_M + NTYPES;
    hl_gemm<<<maxTiles, 256, 0, stream>>>(x, W, bias, ws, out);
}
